// KANHead_35141422416083
// MI455X (gfx1250) — matrix operand order repaired and checked by reading, not compiled, not run on hardware
//
#include <hip/hip_runtime.h>
#include <math.h>

typedef __attribute__((ext_vector_type(16))) __bf16 v16bf;
typedef __attribute__((ext_vector_type(8)))  __bf16 v8bf;
typedef __attribute__((ext_vector_type(8)))  float  v8f;
typedef __attribute__((ext_vector_type(4)))  int    i4;

#define D_IN   1024
#define NB     16
#define BDIM   16384   /* D_IN*NB */
#define BATCH  4096
#define HID    2048
#define LN_EPS 1e-5f

#define BM 128
#define BN 128
#define BK 32
#define LDT 40         /* ushort stride per LDS row = 80 bytes (conflict-free) */
#define NKT (BDIM/BK)  /* 512 */

// ---- CDNA5 async global->LDS path (guarded; falls back to register staging) ----
#if defined(__has_builtin)
#  if __has_builtin(__builtin_amdgcn_global_load_async_to_lds_b128)
#    define HAVE_ASYNC_LDS 1
#  endif
#endif

#if defined(HAVE_ASYNC_LDS)
typedef __attribute__((address_space(1))) i4 g_i4;   // global int4
typedef __attribute__((address_space(3))) i4 l_i4;   // LDS int4
#define ASYNC_B128(gsrc, ldst, off) \
  __builtin_amdgcn_global_load_async_to_lds_b128((g_i4*)(gsrc), (l_i4*)(ldst), (off), 0)
#  if __has_builtin(__builtin_amdgcn_s_wait_asynccnt)
#    define WAIT_ASYNC0() __builtin_amdgcn_s_wait_asynccnt(0)
#  else
#    define WAIT_ASYNC0() asm volatile("s_wait_asynccnt 0x0" ::: "memory")
#  endif
#endif

__device__ __forceinline__ unsigned short f2bf(float f) {
  union { float f; unsigned int u; } v; v.f = f;
  unsigned int r = v.u + 0x7FFFu + ((v.u >> 16) & 1u);   // round-to-nearest-even
  return (unsigned short)(r >> 16);
}

// ---------------------------------------------------------------------------
// Kernel 1: LayerNorm + triangular basis -> z (bf16), and y_lin = x_n . sh_w
// ---------------------------------------------------------------------------
__global__ __launch_bounds__(256) void k_ln_basis(
    const float* __restrict__ x, const float* __restrict__ ln_g,
    const float* __restrict__ ln_b, const float* __restrict__ centers,
    const float* __restrict__ width, const float* __restrict__ sh_w,
    unsigned short* __restrict__ zbf, float* __restrict__ ylin)
{
  const int row = blockIdx.x;
  const int tid = threadIdx.x;
  const float* xr = x + (size_t)row * D_IN;

  float v[4];
  float s = 0.f, s2 = 0.f;
  #pragma unroll
  for (int j = 0; j < 4; ++j) {
    v[j] = xr[tid + j*256];
    s += v[j]; s2 += v[j]*v[j];
  }
  __shared__ float red[2][8];
  #pragma unroll
  for (int o = 16; o > 0; o >>= 1) { s += __shfl_down(s,o); s2 += __shfl_down(s2,o); }
  if ((tid & 31) == 0) { red[0][tid>>5] = s; red[1][tid>>5] = s2; }
  __syncthreads();
  s = 0.f; s2 = 0.f;
  #pragma unroll
  for (int w = 0; w < 8; ++w) { s += red[0][w]; s2 += red[1][w]; }
  const float mu   = s  * (1.f/D_IN);
  const float var  = s2 * (1.f/D_IN) - mu*mu;
  const float rstd = rsqrtf(var + LN_EPS);

  float yl = 0.f;
  #pragma unroll
  for (int j = 0; j < 4; ++j) {
    const int d = tid + j*256;
    const float xn = (v[j]-mu)*rstd*ln_g[d] + ln_b[d];
    yl += xn * sh_w[d];
    const float wd = width[d];
    const float sp = (wd > 20.f) ? wd : log1pf(expf(wd));   // softplus
    const float invw = 1.f / (sp + 1e-6f);
    const float* cd = centers + (size_t)d * NB;
    unsigned int q[8];
    #pragma unroll
    for (int k = 0; k < 8; ++k) {
      float p0 = fmaxf(0.f, 1.f - fabsf((xn - cd[2*k  ]) * invw));
      float p1 = fmaxf(0.f, 1.f - fabsf((xn - cd[2*k+1]) * invw));
      q[k] = (unsigned int)f2bf(p0) | ((unsigned int)f2bf(p1) << 16);
    }
    uint4* dst = (uint4*)(zbf + (size_t)row * BDIM + (size_t)d * NB);
    dst[0] = make_uint4(q[0],q[1],q[2],q[3]);
    dst[1] = make_uint4(q[4],q[5],q[6],q[7]);
  }
  __syncthreads();   // red[] reuse
  #pragma unroll
  for (int o = 16; o > 0; o >>= 1) yl += __shfl_down(yl, o);
  if ((tid & 31) == 0) red[0][tid>>5] = yl;
  __syncthreads();
  if (tid == 0) {
    float t = 0.f;
    #pragma unroll
    for (int w = 0; w < 8; ++w) t += red[0][w];
    ylin[row] = t;
  }
}

// ---------------------------------------------------------------------------
// Kernel 2: fc1_w fp32 -> bf16
// ---------------------------------------------------------------------------
__global__ __launch_bounds__(256) void k_cvt_w(const float* __restrict__ w,
                                               unsigned short* __restrict__ wbf)
{
  const size_t i = ((size_t)blockIdx.x * 256 + threadIdx.x) * 4;
  const float4 f = *(const float4*)(w + i);
  const unsigned int q0 = (unsigned int)f2bf(f.x) | ((unsigned int)f2bf(f.y) << 16);
  const unsigned int q1 = (unsigned int)f2bf(f.z) | ((unsigned int)f2bf(f.w) << 16);
  *(uint2*)(wbf + i) = make_uint2(q0, q1);
}

// ---------------------------------------------------------------------------
// Kernel 3: WMMA GEMM h = gelu(z @ Wt + b), fused rank-1 heads -> partials
// ---------------------------------------------------------------------------
__device__ __forceinline__ v16bf frag16(const unsigned short* p0, const unsigned short* p1) {
  v8bf lo = *(const v8bf*)p0;
  v8bf hi = *(const v8bf*)p1;
  return __builtin_shufflevector(lo, hi, 0,1,2,3,4,5,6,7,8,9,10,11,12,13,14,15);
}

__global__ __launch_bounds__(256) void k_gemm_head(
    const unsigned short* __restrict__ zbf, const unsigned short* __restrict__ wbf,
    const float* __restrict__ fc1_b, const float* __restrict__ hm_w,
    const float* __restrict__ hv_w, float* __restrict__ partM,
    float* __restrict__ partV)
{
  __shared__ unsigned short lA[2][BM*LDT];
  __shared__ unsigned short lB[2][BN*LDT];
  __shared__ float rowM[BM][8];
  __shared__ float rowV[BM][8];

  const int bm = blockIdx.x & 31;   // 32 blocks along M
  const int bn = blockIdx.x >> 5;   // 16 blocks along N
  const int m0 = bm * BM, n0 = bn * BN;
  const int tid  = threadIdx.x;
  const int lane = tid & 31;
  const int wave = tid >> 5;
  const int wm   = wave & 1;        // M half (64 rows)
  const int wn   = wave >> 1;       // N quarter (32 cols)
  const int half = lane >> 4;
  const int l16  = lane & 15;

  const int lrow  = tid >> 1;
  const int lhalf = tid & 1;
  const unsigned short* gA = zbf + (size_t)(m0 + lrow) * BDIM + lhalf * 16;
  const unsigned short* gB = wbf + (size_t)(n0 + lrow) * BDIM + lhalf * 16;
  const int soff = lrow * LDT + lhalf * 16;

  const v8f zero = {0.f,0.f,0.f,0.f,0.f,0.f,0.f,0.f};
  v8f c[4][2];
  #pragma unroll
  for (int mi = 0; mi < 4; ++mi)
    #pragma unroll
    for (int ni = 0; ni < 2; ++ni) c[mi][ni] = zero;

#if defined(HAVE_ASYNC_LDS)
  // ---- async global->LDS pipeline: one barrier per K-step, no VGPR staging ----
  ASYNC_B128(gA, &lA[0][soff], 0);  ASYNC_B128(gA, &lA[0][soff], 16);
  ASYNC_B128(gB, &lB[0][soff], 0);  ASYNC_B128(gB, &lB[0][soff], 16);
  WAIT_ASYNC0();
  __syncthreads();
#else
  uint4 ra0 = ((const uint4*)gA)[0], ra1 = ((const uint4*)gA)[1];
  uint4 rb0 = ((const uint4*)gB)[0], rb1 = ((const uint4*)gB)[1];
  *(uint4*)&lA[0][soff] = ra0; *(uint4*)&lA[0][soff+8] = ra1;
  *(uint4*)&lB[0][soff] = rb0; *(uint4*)&lB[0][soff+8] = rb1;
  __syncthreads();
#endif

  for (int kt = 0; kt < NKT; ++kt) {
    const int cur = kt & 1;
#if defined(HAVE_ASYNC_LDS)
    if (kt + 1 < NKT) {   // kick next tile straight into the other LDS buffer
      const unsigned short* pa = gA + (size_t)(kt+1) * BK;
      const unsigned short* pb = gB + (size_t)(kt+1) * BK;
      unsigned short* la = &lA[cur ^ 1][soff];
      unsigned short* lb = &lB[cur ^ 1][soff];
      ASYNC_B128(pa, la, 0);  ASYNC_B128(pa, la, 16);
      ASYNC_B128(pb, lb, 0);  ASYNC_B128(pb, lb, 16);
    }
#else
    uint4 na0, na1, nb0, nb1;
    if (kt + 1 < NKT) {
      const unsigned short* pa = gA + (size_t)(kt+1) * BK;
      const unsigned short* pb = gB + (size_t)(kt+1) * BK;
      na0 = ((const uint4*)pa)[0]; na1 = ((const uint4*)pa)[1];
      nb0 = ((const uint4*)pb)[0]; nb1 = ((const uint4*)pb)[1];
    }
#endif
    v16bf a[4], b[2];
    #pragma unroll
    for (int mi = 0; mi < 4; ++mi) {
      const unsigned short* p = &lA[cur][(wm*64 + mi*16 + l16) * LDT];
      a[mi] = frag16(p + half*8, p + 16 + half*8);   // K: half*8.. , 16+half*8..
    }
    #pragma unroll
    for (int ni = 0; ni < 2; ++ni) {
      const unsigned short* p = &lB[cur][(wn*32 + ni*16 + l16) * LDT];
      b[ni] = frag16(p + half*8, p + 16 + half*8);   // K: half*8.. , 16+half*8..
    }
    #pragma unroll
    for (int mi = 0; mi < 4; ++mi)
      #pragma unroll
      for (int ni = 0; ni < 2; ++ni)
        c[mi][ni] = __builtin_amdgcn_wmma_f32_16x16x32_bf16(
            false, a[mi], false, b[ni], (short)0, c[mi][ni], false, false);
#if defined(HAVE_ASYNC_LDS)
    if (kt + 1 < NKT) {
      WAIT_ASYNC0();     // my async writes to next buffer have landed in LDS
      __syncthreads();   // everyone's writes have landed; prev-buffer reads done
    }
#else
    __syncthreads();
    if (kt + 1 < NKT) {
      const int nxt = cur ^ 1;
      *(uint4*)&lA[nxt][soff] = na0; *(uint4*)&lA[nxt][soff+8] = na1;
      *(uint4*)&lB[nxt][soff] = nb0; *(uint4*)&lB[nxt][soff+8] = nb1;
      __syncthreads();
    }
#endif
  }

  // Epilogue: bias + exact GELU, dot with hm_w / hv_w, reduce across N-lanes.
  #pragma unroll
  for (int ni = 0; ni < 2; ++ni) {
    const int cc = wn*32 + ni*16 + l16;
    const float bias = fc1_b[n0 + cc];
    const float wmv  = hm_w[n0 + cc];
    const float wvv  = hv_w[n0 + cc];
    #pragma unroll
    for (int mi = 0; mi < 4; ++mi) {
      #pragma unroll
      for (int i = 0; i < 8; ++i) {
        const float vv = c[mi][ni][i] + bias;
        const float h  = 0.5f * vv * (1.f + erff(vv * 0.70710678f));
        float pm = h * wmv;
        float pv = h * wvv;
        #pragma unroll
        for (int sft = 1; sft < 16; sft <<= 1) {   // reduce within each 16-lane half
          pm += __shfl_xor(pm, sft);
          pv += __shfl_xor(pv, sft);
        }
        if (l16 == 0) {
          const int r = wm*64 + mi*16 + i + 8*half;  // C layout: M = i + 8*(lane/16)
          rowM[r][wn*2 + ni] = pm;
          rowV[r][wn*2 + ni] = pv;
        }
      }
    }
  }
  __syncthreads();
  if (tid < BM) {
    float sm = 0.f, sv = 0.f;
    #pragma unroll
    for (int j = 0; j < 8; ++j) { sm += rowM[tid][j]; sv += rowV[tid][j]; }
    partM[(size_t)bn * BATCH + m0 + tid] = sm;
    partV[(size_t)bn * BATCH + m0 + tid] = sv;
  }
}

// ---------------------------------------------------------------------------
// Kernel 4: combine partials + scalar heads
// ---------------------------------------------------------------------------
__global__ __launch_bounds__(256) void k_finalize(
    const float* __restrict__ partM, const float* __restrict__ partV,
    const float* __restrict__ ylin, const float* __restrict__ hm_b,
    const float* __restrict__ hv_b, const float* __restrict__ sh_b,
    const float* __restrict__ gamma, float* __restrict__ out)
{
  const int i = blockIdx.x * 256 + threadIdx.x;
  float sm = 0.f, sv = 0.f;
  #pragma unroll
  for (int j = 0; j < 16; ++j) {
    sm += partM[(size_t)j * BATCH + i];
    sv += partV[(size_t)j * BATCH + i];
  }
  const float g = fmaxf(gamma[0], 0.f);
  out[i]         = sm + hm_b[0] + g * (ylin[i] + sh_b[0]);
  out[BATCH + i] = sv + hv_b[0];
}

// ---------------------------------------------------------------------------
extern "C" void kernel_launch(void* const* d_in, const int* in_sizes, int n_in,
                              void* d_out, int out_size, void* d_ws, size_t ws_size,
                              hipStream_t stream) {
  const float* x       = (const float*)d_in[0];
  const float* ln_g    = (const float*)d_in[1];
  const float* ln_b    = (const float*)d_in[2];
  const float* centers = (const float*)d_in[3];
  const float* width   = (const float*)d_in[4];
  const float* fc1_w   = (const float*)d_in[5];
  const float* fc1_b   = (const float*)d_in[6];
  const float* hm_w    = (const float*)d_in[7];
  const float* hm_b    = (const float*)d_in[8];
  const float* hv_w    = (const float*)d_in[9];
  const float* hv_b    = (const float*)d_in[10];
  const float* sh_w    = (const float*)d_in[11];
  const float* sh_b    = (const float*)d_in[12];
  const float* gamma   = (const float*)d_in[13];

  unsigned short* zbf = (unsigned short*)d_ws;                 // 128 MB
  unsigned short* wbf = zbf + (size_t)BATCH * BDIM;            // +64 MB
  float* ylin  = (float*)(wbf + (size_t)HID * BDIM);
  float* partM = ylin + BATCH;
  float* partV = partM + 16 * BATCH;

  k_ln_basis<<<BATCH, 256, 0, stream>>>(x, ln_g, ln_b, centers, width, sh_w, zbf, ylin);
  k_cvt_w<<<(HID * BDIM) / (256 * 4), 256, 0, stream>>>(fc1_w, wbf);
  k_gemm_head<<<32 * 16, 256, 0, stream>>>(zbf, wbf, fc1_b, hm_w, hv_w, partM, partV);
  k_finalize<<<BATCH / 256, 256, 0, stream>>>(partM, partV, ylin, hm_b, hv_b, sh_b, gamma,
                                              (float*)d_out);
}
